// SLIM_24816321036424
// MI455X (gfx1250) — matrix operand order repaired and checked by reading, not compiled, not run on hardware
//
#include <hip/hip_runtime.h>
#include <hip/hip_bf16.h>

typedef __attribute__((ext_vector_type(16))) _Float16 v16h;
typedef __attribute__((ext_vector_type(8)))  _Float16 v8h;
typedef __attribute__((ext_vector_type(8)))  float    v8f;

#define D_N   50000
#define E_N   500000
#define NT_E  (E_N/16)   // 31250
#define NG_E  (NT_E/2)   // 15625 (M=2 blocking)
#define NT_D  (D_N/16)   // 3125

#define WMMA_F16(a, b, c) \
  __builtin_amdgcn_wmma_f32_16x16x32_f16(false, (a), false, (b), (short)0, (c), false, false)

// ---------- helpers ----------

__device__ __forceinline__ v16h load_pack_f32(const float* __restrict__ p, float sc) {
  // two contiguous 8-float runs at p and p+16, scaled, converted to f16
  float4 a0 = ((const float4*)p)[0];
  float4 a1 = ((const float4*)p)[1];
  float4 b0 = ((const float4*)(p + 16))[0];
  float4 b1 = ((const float4*)(p + 16))[1];
  v16h t;
  t[0]  = (_Float16)(a0.x*sc); t[1]  = (_Float16)(a0.y*sc);
  t[2]  = (_Float16)(a0.z*sc); t[3]  = (_Float16)(a0.w*sc);
  t[4]  = (_Float16)(a1.x*sc); t[5]  = (_Float16)(a1.y*sc);
  t[6]  = (_Float16)(a1.z*sc); t[7]  = (_Float16)(a1.w*sc);
  t[8]  = (_Float16)(b0.x*sc); t[9]  = (_Float16)(b0.y*sc);
  t[10] = (_Float16)(b0.z*sc); t[11] = (_Float16)(b0.w*sc);
  t[12] = (_Float16)(b1.x*sc); t[13] = (_Float16)(b1.y*sc);
  t[14] = (_Float16)(b1.z*sc); t[15] = (_Float16)(b1.w*sc);
  return t;
}

__device__ __forceinline__ v16h load_pack_h(const _Float16* __restrict__ p) {
  v8h lo = *(const v8h*)p;
  v8h hi = *(const v8h*)(p + 16);
  v16h t;
  #pragma unroll
  for (int e = 0; e < 8; ++e) { t[e] = lo[e]; t[8 + e] = hi[e]; }
  return t;
}

// ---------- repack: fp32 [Kin x 128] row-major -> f16 B-fragment layout ----------
// fragment (kt,nt) occupies 512 halves; element index = lane*16 + e
//   N = nt*16 + (lane&15);  K = kt*32 + ((e<8)?e:e+8) + ((lane<16)?0:8)
__global__ void repack_w_kernel(const float* __restrict__ W, _Float16* __restrict__ out,
                                int Kin, int KT) {
  int idx = blockIdx.x * blockDim.x + threadIdx.x;
  int total = KT * 8 * 512;
  if (idx >= total) return;
  int frag   = idx >> 9;
  int within = idx & 511;
  int lane   = within >> 4;
  int e      = within & 15;
  int kt = frag >> 3;
  int nt = frag & 7;
  int n  = nt * 16 + (lane & 15);
  int k  = kt * 32 + ((e < 8) ? e : e + 8) + ((lane < 16) ? 0 : 8);
  float v = (k < Kin) ? W[(size_t)k * 128 + n] : 0.0f;
  out[idx] = (_Float16)v;
}

// ---------- zero ----------
__global__ void zero_kernel(float* __restrict__ p, size_t n) {
  size_t i = (size_t)blockIdx.x * blockDim.x + threadIdx.x;
  size_t stride = (size_t)gridDim.x * blockDim.x;
  for (; i < n; i += stride) p[i] = 0.0f;
}

// ---------- degree count ----------
__global__ void count_kernel(const int* __restrict__ dst_idx, float* __restrict__ cnt) {
  int e = blockIdx.x * blockDim.x + threadIdx.x;
  if (e < E_N) atomicAdd(cnt + dst_idx[e], 1.0f);
}

// ---------- edge MLP + scatter (M=2 blocking: one wave = 32 edges) ----------
// LDS: [0,160KB) weights (w1:49152h, w2:16384h, w3:16384h),
//      [160KB,224KB) 8 waves x 2 tiles x 2048h stage, [224KB,+400B) time table
__global__ void __launch_bounds__(256, 1) edge_kernel(
    const float* __restrict__ h_src, const float* __restrict__ edge_f,
    const float* __restrict__ dtv,   const float* __restrict__ edge_w,
    const int*   __restrict__ dst_idx,
    const _Float16* __restrict__ wfrag,
    const float* __restrict__ b1, const float* __restrict__ b2, const float* __restrict__ b3,
    float* __restrict__ s_out)
{
  extern __shared__ char smem[];
  _Float16* wlds      = (_Float16*)smem;               // 81920 halves
  _Float16* stage_all = wlds + 81920;                  // 16 * 2048 halves
  float*    twl       = (float*)(smem + 229376);       // 100 floats

  {
    const uint4* src = (const uint4*)wfrag;
    uint4* dst = (uint4*)wlds;
    for (int i = threadIdx.x; i < 10240; i += 256) dst[i] = src[i];
    if (threadIdx.x < 100)
      twl[threadIdx.x] = __powf(10.0f, -(9.0f / 99.0f) * (float)threadIdx.x);
  }
  __syncthreads();

  _Float16* w1s = wlds;
  _Float16* w2s = wlds + 49152;
  _Float16* w3s = wlds + 65536;

  const int wave  = threadIdx.x >> 5;
  const int lane  = threadIdx.x & 31;
  const int row   = lane & 15;
  const int h8    = (lane >> 4) << 3;   // 0 or 8
  const int kboff = h8;                 // A-layout K base
  _Float16* stage0 = stage_all + wave * 4096;
  _Float16* stage1 = stage0 + 2048;

  for (int g = blockIdx.x * 8 + wave; g < NG_E; g += gridDim.x * 8) {
    const int m0 = 32 * g + row;        // tile 2g
    const int m1 = m0 + 16;             // tile 2g+1
    v8f acc0[8], acc1[8];

    // ================= layer 1: 16x384 @ 384x128 =================
    #pragma unroll
    for (int nt = 0; nt < 8; ++nt) {
      float bv = b1[nt * 16 + row];
      #pragma unroll
      for (int i = 0; i < 8; ++i) { acc0[nt][i] = bv; acc1[nt][i] = bv; }
    }
    const float* hs0 = h_src + (size_t)(D_N + m0) * 128 + kboff;
    const float* hs1 = h_src + (size_t)(D_N + m1) * 128 + kboff;
    const float* ef0 = edge_f + (size_t)m0 * 128 + kboff;
    const float* ef1 = edge_f + (size_t)m1 * 128 + kboff;
    #pragma unroll 1
    for (int kt = 0; kt < 8; ++kt) {
      const float* p0 = (kt < 4) ? (hs0 + kt * 32) : (ef0 + (kt - 4) * 32);
      const float* p1 = (kt < 4) ? (hs1 + kt * 32) : (ef1 + (kt - 4) * 32);
      v16h af0 = load_pack_f32(p0, 1.0f);
      v16h af1 = load_pack_f32(p1, 1.0f);
      #pragma unroll
      for (int nt = 0; nt < 8; ++nt) {
        v16h bf = *(const v16h*)(w1s + (((kt << 3) + nt) << 9) + (lane << 4));
        acc0[nt] = WMMA_F16(af0, bf, acc0[nt]);
        acc1[nt] = WMMA_F16(af1, bf, acc1[nt]);
      }
    }
    const float tv0 = dtv[m0];
    const float tv1 = dtv[m1];
    #pragma unroll 1
    for (int kt = 8; kt < 12; ++kt) {
      v16h af0, af1;
      #pragma unroll
      for (int e = 0; e < 16; ++e) {
        int klocal = kboff + (e < 8 ? e : 8 + e);   // interleaved A K-mapping
        int td = (kt - 8) * 32 + klocal;
        float w = (td < 100) ? twl[td] : 0.0f;
        af0[e] = (_Float16)((td < 100) ? __cosf(tv0 * w) : 0.0f);
        af1[e] = (_Float16)((td < 100) ? __cosf(tv1 * w) : 0.0f);
      }
      #pragma unroll
      for (int nt = 0; nt < 8; ++nt) {
        v16h bf = *(const v16h*)(w1s + (((kt << 3) + nt) << 9) + (lane << 4));
        acc0[nt] = WMMA_F16(af0, bf, acc0[nt]);
        acc1[nt] = WMMA_F16(af1, bf, acc1[nt]);
      }
    }
    #pragma unroll
    for (int nt = 0; nt < 8; ++nt)
      #pragma unroll
      for (int r = 0; r < 8; ++r) {
        stage0[(r + h8) * 128 + nt * 16 + row] = (_Float16)fmaxf(acc0[nt][r], 0.0f);
        stage1[(r + h8) * 128 + nt * 16 + row] = (_Float16)fmaxf(acc1[nt][r], 0.0f);
      }

    // ================= layer 2: 16x128 @ 128x128 =================
    #pragma unroll
    for (int nt = 0; nt < 8; ++nt) {
      float bv = b2[nt * 16 + row];
      #pragma unroll
      for (int i = 0; i < 8; ++i) { acc0[nt][i] = bv; acc1[nt][i] = bv; }
    }
    #pragma unroll 1
    for (int kt = 0; kt < 4; ++kt) {
      v16h af0 = load_pack_h(stage0 + row * 128 + kt * 32 + kboff);
      v16h af1 = load_pack_h(stage1 + row * 128 + kt * 32 + kboff);
      #pragma unroll
      for (int nt = 0; nt < 8; ++nt) {
        v16h bf = *(const v16h*)(w2s + (((kt << 3) + nt) << 9) + (lane << 4));
        acc0[nt] = WMMA_F16(af0, bf, acc0[nt]);
        acc1[nt] = WMMA_F16(af1, bf, acc1[nt]);
      }
    }
    #pragma unroll
    for (int nt = 0; nt < 8; ++nt)
      #pragma unroll
      for (int r = 0; r < 8; ++r) {
        stage0[(r + h8) * 128 + nt * 16 + row] = (_Float16)fmaxf(acc0[nt][r], 0.0f);
        stage1[(r + h8) * 128 + nt * 16 + row] = (_Float16)fmaxf(acc1[nt][r], 0.0f);
      }

    // ================= layer 3: 16x128 @ 128x128, *edge_w, scatter =================
    #pragma unroll
    for (int nt = 0; nt < 8; ++nt) {
      float bv = b3[nt * 16 + row];
      #pragma unroll
      for (int i = 0; i < 8; ++i) { acc0[nt][i] = bv; acc1[nt][i] = bv; }
    }
    #pragma unroll 1
    for (int kt = 0; kt < 4; ++kt) {
      v16h af0 = load_pack_h(stage0 + row * 128 + kt * 32 + kboff);
      v16h af1 = load_pack_h(stage1 + row * 128 + kt * 32 + kboff);
      #pragma unroll
      for (int nt = 0; nt < 8; ++nt) {
        v16h bf = *(const v16h*)(w3s + (((kt << 3) + nt) << 9) + (lane << 4));
        acc0[nt] = WMMA_F16(af0, bf, acc0[nt]);
        acc1[nt] = WMMA_F16(af1, bf, acc1[nt]);
      }
    }
    float ew0[8], ew1[8];
    int   di0[8], di1[8];
    #pragma unroll
    for (int r = 0; r < 8; ++r) {
      int mr0 = 32 * g + r + h8;
      int mr1 = mr0 + 16;
      ew0[r] = edge_w[mr0];  di0[r] = dst_idx[mr0];
      ew1[r] = edge_w[mr1];  di1[r] = dst_idx[mr1];
    }
    #pragma unroll 1
    for (int nt = 0; nt < 8; ++nt) {
      #pragma unroll
      for (int r = 0; r < 8; ++r) {
        atomicAdd(s_out + (size_t)di0[r] * 128 + nt * 16 + row, acc0[nt][r] * ew0[r]);
        atomicAdd(s_out + (size_t)di1[r] * 128 + nt * 16 + row, acc1[nt][r] * ew1[r]);
      }
    }
  }
}

// ---------- dst combine MLP + LayerNorm ----------
// LDS: [0,128KB) weights (c1:32768h, c2:16384h, c3:16384h), then 8 * 8KB per-wave stage
__global__ void __launch_bounds__(256, 1) dst_kernel(
    const float* __restrict__ h_src, const float* __restrict__ s_in,
    const float* __restrict__ cntv,
    const _Float16* __restrict__ cfrag,
    const float* __restrict__ c1b, const float* __restrict__ c2b, const float* __restrict__ c3b,
    const float* __restrict__ lng, const float* __restrict__ lnb,
    const float* __restrict__ ln2g, const float* __restrict__ ln2b,
    float* __restrict__ out)
{
  extern __shared__ char smem[];
  _Float16* wlds = (_Float16*)smem;    // 65536 halves
  {
    const uint4* src = (const uint4*)cfrag;
    uint4* dst = (uint4*)wlds;
    for (int i = threadIdx.x; i < 8192; i += 256) dst[i] = src[i];
  }
  __syncthreads();
  _Float16* c1s = wlds;
  _Float16* c2s = wlds + 32768;
  _Float16* c3s = wlds + 49152;

  const int wave  = threadIdx.x >> 5;
  const int lane  = threadIdx.x & 31;
  const int row   = lane & 15;
  const int h8    = (lane >> 4) << 3;
  const int kboff = h8;
  char* wreg = smem + 131072 + wave * 8192;
  _Float16* stage  = (_Float16*)wreg;   // 16x128 f16 (layers 1-2)
  float*    stageF = (float*)wreg;      // 16x128 f32 (layer 3 / LN), time-disjoint

  float4 g1  = ((const float4*)lng)[lane];
  float4 bb1 = ((const float4*)lnb)[lane];
  float4 g2  = ((const float4*)ln2g)[lane];
  float4 bb2 = ((const float4*)ln2b)[lane];

  for (int tile = blockIdx.x * 8 + wave; tile < NT_D; tile += gridDim.x * 8) {
    const int dn = tile * 16 + row;
    const float inv = 1.0f / fmaxf(cntv[dn], 1.0f);
    v8f acc[8];

    // ---- layer 1: 16x256 @ 256x128 ----
    #pragma unroll
    for (int nt = 0; nt < 8; ++nt) {
      float bv = c1b[nt * 16 + row];
      #pragma unroll
      for (int i = 0; i < 8; ++i) acc[nt][i] = bv;
    }
    const float* sp = s_in + (size_t)dn * 128 + kboff;
    const float* hp = h_src + (size_t)dn * 128 + kboff;
    #pragma unroll 1
    for (int kt = 0; kt < 8; ++kt) {
      const float* p = (kt < 4) ? (sp + kt * 32) : (hp + (kt - 4) * 32);
      float sc = (kt < 4) ? inv : 1.0f;
      v16h af = load_pack_f32(p, sc);
      #pragma unroll
      for (int nt = 0; nt < 8; ++nt) {
        v16h bf = *(const v16h*)(c1s + (((kt << 3) + nt) << 9) + (lane << 4));
        acc[nt] = WMMA_F16(af, bf, acc[nt]);
      }
    }
    #pragma unroll
    for (int nt = 0; nt < 8; ++nt)
      #pragma unroll
      for (int r = 0; r < 8; ++r)
        stage[(r + h8) * 128 + nt * 16 + row] = (_Float16)fmaxf(acc[nt][r], 0.0f);

    // ---- layer 2 ----
    #pragma unroll
    for (int nt = 0; nt < 8; ++nt) {
      float bv = c2b[nt * 16 + row];
      #pragma unroll
      for (int i = 0; i < 8; ++i) acc[nt][i] = bv;
    }
    #pragma unroll 1
    for (int kt = 0; kt < 4; ++kt) {
      v16h af = load_pack_h(stage + row * 128 + kt * 32 + kboff);
      #pragma unroll
      for (int nt = 0; nt < 8; ++nt) {
        v16h bf = *(const v16h*)(c2s + (((kt << 3) + nt) << 9) + (lane << 4));
        acc[nt] = WMMA_F16(af, bf, acc[nt]);
      }
    }
    #pragma unroll
    for (int nt = 0; nt < 8; ++nt)
      #pragma unroll
      for (int r = 0; r < 8; ++r)
        stage[(r + h8) * 128 + nt * 16 + row] = (_Float16)fmaxf(acc[nt][r], 0.0f);

    // ---- layer 3 -> stageF (f32) ----
    #pragma unroll
    for (int nt = 0; nt < 8; ++nt) {
      float bv = c3b[nt * 16 + row];
      #pragma unroll
      for (int i = 0; i < 8; ++i) acc[nt][i] = bv;
    }
    #pragma unroll 1
    for (int kt = 0; kt < 4; ++kt) {
      v16h af = load_pack_h(stage + row * 128 + kt * 32 + kboff);
      #pragma unroll
      for (int nt = 0; nt < 8; ++nt) {
        v16h bf = *(const v16h*)(c3s + (((kt << 3) + nt) << 9) + (lane << 4));
        acc[nt] = WMMA_F16(af, bf, acc[nt]);
      }
    }
    #pragma unroll
    for (int nt = 0; nt < 8; ++nt)
      #pragma unroll
      for (int r = 0; r < 8; ++r)
        stageF[(r + h8) * 128 + nt * 16 + row] = acc[nt][r];

    // ---- LayerNorm(tgt) + LayerNorm(s): whole wave cooperates ----
    #pragma unroll 1
    for (int mr = 0; mr < 16; ++mr) {
      float4 tg = ((const float4*)(stageF + mr * 128))[lane];
      const float* srow = s_in + (size_t)(tile * 16 + mr) * 128;
      float4 sv = ((const float4*)srow)[lane];
      float st  = tg.x + tg.y + tg.z + tg.w;
      float st2 = tg.x*tg.x + tg.y*tg.y + tg.z*tg.z + tg.w*tg.w;
      float ss  = sv.x + sv.y + sv.z + sv.w;
      float ss2 = sv.x*sv.x + sv.y*sv.y + sv.z*sv.z + sv.w*sv.w;
      #pragma unroll
      for (int off = 16; off > 0; off >>= 1) {
        st  += __shfl_xor(st,  off, 32);
        st2 += __shfl_xor(st2, off, 32);
        ss  += __shfl_xor(ss,  off, 32);
        ss2 += __shfl_xor(ss2, off, 32);
      }
      float mt = st * (1.0f / 128.0f);
      float vt = st2 * (1.0f / 128.0f) - mt * mt;
      float ms = ss * (1.0f / 128.0f);
      float vs = ss2 * (1.0f / 128.0f) - ms * ms;
      float rt = rsqrtf(vt + 1e-5f);
      float rs = rsqrtf(vs + 1e-5f);
      float4 o;
      o.x = (tg.x - mt) * rt * g1.x + bb1.x + (sv.x - ms) * rs * g2.x + bb2.x;
      o.y = (tg.y - mt) * rt * g1.y + bb1.y + (sv.y - ms) * rs * g2.y + bb2.y;
      o.z = (tg.z - mt) * rt * g1.z + bb1.z + (sv.z - ms) * rs * g2.z + bb2.z;
      o.w = (tg.w - mt) * rt * g1.w + bb1.w + (sv.w - ms) * rs * g2.w + bb2.w;
      ((float4*)(out + (size_t)(tile * 16 + mr) * 128))[lane] = o;
    }
  }
}

// ---------- launcher ----------
extern "C" void kernel_launch(void* const* d_in, const int* in_sizes, int n_in,
                              void* d_out, int out_size, void* d_ws, size_t ws_size,
                              hipStream_t stream) {
  (void)in_sizes; (void)n_in; (void)out_size; (void)ws_size;
  const float* h_src  = (const float*)d_in[0];
  const float* edge_f = (const float*)d_in[1];
  const float* dtv    = (const float*)d_in[2];
  const float* edge_w = (const float*)d_in[3];
  const int*   dst_i  = (const int*)d_in[4];
  const float* W1 = (const float*)d_in[5];  const float* b1 = (const float*)d_in[6];
  const float* W2 = (const float*)d_in[7];  const float* b2 = (const float*)d_in[8];
  const float* W3 = (const float*)d_in[9];  const float* b3 = (const float*)d_in[10];
  const float* C1 = (const float*)d_in[11]; const float* c1 = (const float*)d_in[12];
  const float* C2 = (const float*)d_in[13]; const float* c2 = (const float*)d_in[14];
  const float* C3 = (const float*)d_in[15]; const float* c3 = (const float*)d_in[16];
  const float* lng  = (const float*)d_in[17]; const float* lnb  = (const float*)d_in[18];
  const float* ln2g = (const float*)d_in[19]; const float* ln2b = (const float*)d_in[20];

  float* s   = (float*)d_ws;                       // D*128 f32
  float* cnt = s + (size_t)D_N * 128;              // D f32
  _Float16* wf  = (_Float16*)(cnt + D_N);          // repacked weights (f16)
  _Float16* w1f = wf;                              // 49152 halves
  _Float16* w2f = wf + 49152;                      // 16384
  _Float16* w3f = wf + 65536;                      // 16384
  _Float16* cf  = wf + 81920;
  _Float16* c1f = cf;                              // 32768
  _Float16* c2f = cf + 32768;                      // 16384
  _Float16* c3f = cf + 49152;                      // 16384

  zero_kernel<<<2048, 256, 0, stream>>>(s, (size_t)D_N * 128 + D_N);
  repack_w_kernel<<<192, 256, 0, stream>>>(W1, w1f, 356, 12);
  repack_w_kernel<<< 64, 256, 0, stream>>>(W2, w2f, 128, 4);
  repack_w_kernel<<< 64, 256, 0, stream>>>(W3, w3f, 128, 4);
  repack_w_kernel<<<128, 256, 0, stream>>>(C1, c1f, 256, 8);
  repack_w_kernel<<< 64, 256, 0, stream>>>(C2, c2f, 128, 4);
  repack_w_kernel<<< 64, 256, 0, stream>>>(C3, c3f, 128, 4);
  count_kernel<<<(E_N + 255) / 256, 256, 0, stream>>>(dst_i, cnt);

  size_t shmem_e = 163840 + 65536 + 512;   // weights + 16 stages + time table
  edge_kernel<<<1024, 256, shmem_e, stream>>>(h_src, edge_f, dtv, edge_w, dst_i,
                                              w1f, b1, b2, b3, s);
  size_t shmem_d = 131072 + 65536;         // weights + per-wave stages
  dst_kernel<<<512, 256, shmem_d, stream>>>(h_src, s, cnt, c1f, c1, c2, c3,
                                            lng, lnb, ln2g, ln2b, (float*)d_out);
}
